// CausalSelfAttention_29892972380599
// MI455X (gfx1250) — compile-verified, matrix operand order checked by reading, not run on hardware
//
#include <hip/hip_runtime.h>

#define BB  4
#define SS  2048
#define DD  2048
#define HH  16
#define DKK 128
#define MM  (BB*SS)   // 8192 rows for every GEMM

typedef __attribute__((ext_vector_type(16))) __bf16 v16bf;
typedef __attribute__((ext_vector_type(8)))  float  v8f;

union Frag { v16bf v; uint4 q[2]; };
union Acc  { v8f  v; float f[8]; };

static __device__ __forceinline__ unsigned short f2bf(float x) {
  unsigned int u = __float_as_uint(x);
  u += 0x7FFFu + ((u >> 16) & 1u);          // round-to-nearest-even
  return (unsigned short)(u >> 16);
}

static __device__ __forceinline__ v8f wmma_bf16(const Frag& a, const Frag& b, v8f c) {
  // D = A(16x32) x B(32x16) + C, f32 accumulate
  return __builtin_amdgcn_wmma_f32_16x16x32_bf16(
      /*neg_a=*/false, a.v, /*neg_b=*/false, b.v,
      /*c_mod=*/(short)0, c, /*reuse_a=*/false, /*reuse_b=*/false);
}

// CDNA5 async Global->LDS copy (16 bytes per lane), tracked by ASYNCcnt.
static __device__ __forceinline__ void async_ld_b128(unsigned lds_addr, const void* gaddr) {
  asm volatile("global_load_async_to_lds_b128 %0, %1, off"
               :: "v"(lds_addr), "v"((unsigned long long)(size_t)gaddr)
               : "memory");
}
static __device__ __forceinline__ void wait_asynccnt0() {
  asm volatile("s_wait_asynccnt 0x0" ::: "memory");
}

// ---------------------------------------------------------------- convert
__global__ __launch_bounds__(256)
void cvt_f32_to_bf16(const float* __restrict__ src, unsigned short* __restrict__ dst, int n) {
  int i = blockIdx.x * blockDim.x + threadIdx.x;
  if (i < n) dst[i] = f2bf(src[i]);
}

// ---------------------------------------------------------------- GEMM
// Y[m,n] = sum_k A[m,k] * W[n,k] + bias[n]
// A: MM x DD (bf16, K-major), W: DD x DD (bf16, K-major)
// Tiles: BM=128, BN=128, BK=32; 8 waves (4x2), each wave 32x64 (2x4 WMMA frags).
// Double-buffered LDS fed by global_load_async_to_lds_b128 (ASYNCcnt).
// mode 0: bf16 row-major out ; mode 1: bf16 V-transposed (B,D,S) ; mode 2: f32 row-major out
__global__ __launch_bounds__(256)
void gemm_bf16_wmma(const unsigned short* __restrict__ A,
                    const unsigned short* __restrict__ W,
                    const float* __restrict__ bias,
                    void* __restrict__ out, int mode)
{
  __shared__ unsigned short As[2][128 * 32];
  __shared__ unsigned short Bs[2][128 * 32];

  const int tid  = threadIdx.x;
  const int w    = tid >> 5, lane = tid & 31;
  const int wm   = w >> 1,   wn   = w & 1;            // 4x2 wave grid -> 32x64 per wave
  const int bm   = blockIdx.y * 128, bn = blockIdx.x * 128;
  const int fr   = lane & 15;
  const int ko   = (lane >> 4) << 3;
  const int lgrp = lane >> 4, ncol = lane & 15;

  Acc acc[2][4];
#pragma unroll
  for (int i = 0; i < 2; ++i)
#pragma unroll
    for (int j = 0; j < 4; ++j)
#pragma unroll
      for (int e = 0; e < 8; ++e) acc[i][j].f[e] = 0.f;

  // tile copy mapping: 128 rows x 32 cols bf16, 256 threads, 32B each
  const int ra = tid >> 1, ca = (tid & 1) * 16;

  // issue async copies for the k0 tile into buffer `buf`
  auto issue_tile = [&](int buf, int k0) {
    const unsigned short* ga = A + (size_t)(bm + ra) * DD + k0 + ca;
    async_ld_b128((unsigned)(size_t)&As[buf][ra * 32 + ca],     ga);
    async_ld_b128((unsigned)(size_t)&As[buf][ra * 32 + ca + 8], ga + 8);
    const unsigned short* gb = W + (size_t)(bn + ra) * DD + k0 + ca;
    async_ld_b128((unsigned)(size_t)&Bs[buf][ra * 32 + ca],     gb);
    async_ld_b128((unsigned)(size_t)&Bs[buf][ra * 32 + ca + 8], gb + 8);
  };

  issue_tile(0, 0);

  for (int k0 = 0; k0 < DD; k0 += 32) {
    const int cur = (k0 >> 5) & 1;
    wait_asynccnt0();       // this wave's copies into buf `cur` are done
    __syncthreads();        // everyone's copies visible; prev readers of buf cur^1 done
    if (k0 + 32 < DD) issue_tile(cur ^ 1, k0 + 32);

    Frag a[2], bfr[4];
    const unsigned short* ar = &As[cur][(wm * 32 + fr) * 32 + ko];
#pragma unroll
    for (int i = 0; i < 2; ++i) {
      a[i].q[0] = *(const uint4*)(ar + i * 512);
      a[i].q[1] = *(const uint4*)(ar + i * 512 + 16);
    }
    const unsigned short* br = &Bs[cur][(wn * 64 + fr) * 32 + ko];
#pragma unroll
    for (int i = 0; i < 4; ++i) {
      bfr[i].q[0] = *(const uint4*)(br + i * 512);
      bfr[i].q[1] = *(const uint4*)(br + i * 512 + 16);
    }
#pragma unroll
    for (int fm = 0; fm < 2; ++fm)
#pragma unroll
      for (int fn = 0; fn < 4; ++fn)
        acc[fm][fn].v = wmma_bf16(a[fm], bfr[fn], acc[fm][fn].v);

    __syncthreads();        // all waves done reading buf `cur` before it is refilled
  }

#pragma unroll
  for (int fm = 0; fm < 2; ++fm)
#pragma unroll
    for (int fn = 0; fn < 4; ++fn) {
      const int gn = bn + wn * 64 + fn * 16 + ncol;
      const float bv = bias[gn];
#pragma unroll
      for (int j = 0; j < 8; ++j) {
        const int gm = bm + wm * 32 + fm * 16 + j + 8 * lgrp;
        const float v = acc[fm][fn].f[j] + bv;
        if (mode == 2) {
          ((float*)out)[(size_t)gm * DD + gn] = v;
        } else if (mode == 0) {
          ((unsigned short*)out)[(size_t)gm * DD + gn] = f2bf(v);
        } else {                                   // V transposed: (b, n, s)
          const int b = gm >> 11, s = gm & (SS - 1);
          ((unsigned short*)out)[((size_t)b * DD + gn) * SS + s] = f2bf(v);
        }
      }
    }
}

// ---------------------------------------------------------------- flash attention
// Q,K: (B,S,D) bf16 (head slices contiguous, ld=D). Vt: (B,D,S) bf16 (ld=S).
// One wave == 16 query rows; 8 waves/block; online softmax; causal.
__global__ __launch_bounds__(256)
void attn_fwd_wmma(const unsigned short* __restrict__ Q,
                   const unsigned short* __restrict__ Kq,
                   const unsigned short* __restrict__ Vt,
                   unsigned short* __restrict__ Ctx)
{
  __shared__ unsigned short Plds[8][16 * 32];   // per-wave P tile (no barriers needed)

  const int tid  = threadIdx.x, w = tid >> 5, lane = tid & 31;
  const int b    = blockIdx.z, h = blockIdx.y;
  const int q0   = blockIdx.x * 128 + w * 16;
  const int fr   = lane & 15;
  const int ko   = (lane >> 4) << 3;
  const int lgrp = lane >> 4, ncol = lane & 15;
  const float sc_log2e = 1.44269504088896f / 11.3137084989848f;  // log2(e)/sqrt(128)

  // Q fragments for all 4 dk-steps (K = 128 = 4 x 32)
  Frag qf[4];
  const unsigned short* qrow = Q + ((size_t)b * SS + q0 + fr) * DD + h * DKK;
#pragma unroll
  for (int ks = 0; ks < 4; ++ks) {
    qf[ks].q[0] = *(const uint4*)(qrow + ks * 32 + ko);
    qf[ks].q[1] = *(const uint4*)(qrow + ks * 32 + ko + 16);
  }

  Acc o[8];
#pragma unroll
  for (int t = 0; t < 8; ++t)
#pragma unroll
    for (int e = 0; e < 8; ++e) o[t].f[e] = 0.f;
  float rmax[8], rsum[8];
#pragma unroll
  for (int j = 0; j < 8; ++j) { rmax[j] = -3.0e38f; rsum[j] = 0.f; }

  unsigned short* pl = Plds[w];

  for (int kv0 = 0; kv0 < q0 + 16; kv0 += 32) {
    // ---- S = Q K^T (two 16x16 tiles covering kv0..kv0+31)
    Acc s0, s1;
#pragma unroll
    for (int e = 0; e < 8; ++e) { s0.f[e] = 0.f; s1.f[e] = 0.f; }
#pragma unroll
    for (int ks = 0; ks < 4; ++ks) {
      Frag kf;
      const unsigned short* kr = Kq + ((size_t)b * SS + kv0 + fr) * DD + h * DKK + ks * 32 + ko;
      kf.q[0] = *(const uint4*)kr;            kf.q[1] = *(const uint4*)(kr + 16);
      s0.v = wmma_bf16(qf[ks], kf, s0.v);
      const unsigned short* kr2 = kr + 16 * DD;
      kf.q[0] = *(const uint4*)kr2;           kf.q[1] = *(const uint4*)(kr2 + 16);
      s1.v = wmma_bf16(qf[ks], kf, s1.v);
    }

    // ---- causal mask + online softmax (log2-domain), P -> LDS in row-major
    float cf[8];
#pragma unroll
    for (int j = 0; j < 8; ++j) {
      const int qr  = q0 + j + 8 * lgrp;
      const int kva = kv0 + ncol, kvb = kv0 + 16 + ncol;
      float sa = (kva <= qr) ? s0.f[j] * sc_log2e : -3.0e38f;
      float sb = (kvb <= qr) ? s1.f[j] * sc_log2e : -3.0e38f;
      float pm = fmaxf(sa, sb);
#pragma unroll
      for (int msk = 1; msk <= 8; msk <<= 1) pm = fmaxf(pm, __shfl_xor(pm, msk, 32));
      const float mnew = fmaxf(rmax[j], pm);
      cf[j]   = exp2f(rmax[j] - mnew);
      rmax[j] = mnew;
      const float pa = exp2f(sa - mnew), pb = exp2f(sb - mnew);
      float ps = pa + pb;
#pragma unroll
      for (int msk = 1; msk <= 8; msk <<= 1) ps += __shfl_xor(ps, msk, 32);
      rsum[j] = rsum[j] * cf[j] + ps;
      const int m = j + 8 * lgrp;
      pl[m * 32 + ncol]      = f2bf(pa);
      pl[m * 32 + 16 + ncol] = f2bf(pb);
    }

    // ---- reload P as A-fragment (16x32), same-wave LDS ops are in-order
    Frag pf;
    pf.q[0] = *(const uint4*)&pl[fr * 32 + ko];
    pf.q[1] = *(const uint4*)&pl[fr * 32 + ko + 16];

    // ---- O = diag(cf)*O + P V  (8 dk tiles of 16)
#pragma unroll
    for (int t = 0; t < 8; ++t) {
      Frag vf;
      const unsigned short* vr = Vt + (size_t)b * DD * SS
                               + (size_t)(h * DKK + t * 16 + fr) * SS + kv0 + ko;
      vf.q[0] = *(const uint4*)vr;  vf.q[1] = *(const uint4*)(vr + 16);
#pragma unroll
      for (int j = 0; j < 8; ++j) o[t].f[j] *= cf[j];
      o[t].v = wmma_bf16(pf, vf, o[t].v);
    }
  }

  // ---- normalize + store ctx (B,S,D) bf16
  float inv[8];
#pragma unroll
  for (int j = 0; j < 8; ++j) inv[j] = 1.0f / rsum[j];
#pragma unroll
  for (int t = 0; t < 8; ++t)
#pragma unroll
    for (int j = 0; j < 8; ++j) {
      const int m = q0 + j + 8 * lgrp;
      Ctx[((size_t)b * SS + m) * DD + h * DKK + t * 16 + ncol] = f2bf(o[t].f[j] * inv[j]);
    }
}

// ---------------------------------------------------------------- launch
extern "C" void kernel_launch(void* const* d_in, const int* in_sizes, int n_in,
                              void* d_out, int out_size, void* d_ws, size_t ws_size,
                              hipStream_t stream) {
  (void)in_sizes; (void)n_in; (void)out_size; (void)ws_size;
  const float* x  = (const float*)d_in[0];
  const float* Wq = (const float*)d_in[1];
  const float* bq = (const float*)d_in[2];
  const float* Wk = (const float*)d_in[3];
  const float* bk = (const float*)d_in[4];
  const float* Wv = (const float*)d_in[5];
  const float* bv = (const float*)d_in[6];
  const float* Wo = (const float*)d_in[7];
  const float* bo = (const float*)d_in[8];

  char* ws = (char*)d_ws;                 // 160 MiB total
  unsigned short* xb   = (unsigned short*)(ws);                       // 32 MiB
  unsigned short* Wqb  = (unsigned short*)(ws + (size_t)33554432);    // 8 MiB
  unsigned short* Wkb  = (unsigned short*)(ws + (size_t)41943040);
  unsigned short* Wvb  = (unsigned short*)(ws + (size_t)50331648);
  unsigned short* Wob  = (unsigned short*)(ws + (size_t)58720256);
  unsigned short* Qb   = (unsigned short*)(ws + (size_t)67108864);    // 32 MiB
  unsigned short* Kb   = (unsigned short*)(ws + (size_t)100663296);   // 32 MiB
  unsigned short* Vtb  = (unsigned short*)(ws + (size_t)134217728);   // 32 MiB
  unsigned short* Ctxb = xb;              // reuse x-bf16 buffer after projections

  const int NX = BB * SS * DD;            // 16,777,216
  const int NW = DD * DD;                 //  4,194,304
  cvt_f32_to_bf16<<<(NX + 255) / 256, 256, 0, stream>>>(x,  xb,  NX);
  cvt_f32_to_bf16<<<(NW + 255) / 256, 256, 0, stream>>>(Wq, Wqb, NW);
  cvt_f32_to_bf16<<<(NW + 255) / 256, 256, 0, stream>>>(Wk, Wkb, NW);
  cvt_f32_to_bf16<<<(NW + 255) / 256, 256, 0, stream>>>(Wv, Wvb, NW);
  cvt_f32_to_bf16<<<(NW + 255) / 256, 256, 0, stream>>>(Wo, Wob, NW);

  dim3 gg(DD / 128, MM / 128);            // (16, 64)
  gemm_bf16_wmma<<<gg, 256, 0, stream>>>(xb, Wqb, bq, (void*)Qb, 0);
  gemm_bf16_wmma<<<gg, 256, 0, stream>>>(xb, Wkb, bk, (void*)Kb, 0);
  gemm_bf16_wmma<<<gg, 256, 0, stream>>>(xb, Wvb, bv, (void*)Vtb, 1);

  attn_fwd_wmma<<<dim3(SS / 128, HH, BB), 256, 0, stream>>>(Qb, Kb, Vtb, Ctxb);

  gemm_bf16_wmma<<<gg, 256, 0, stream>>>(Ctxb, Wob, bo, d_out, 2);
}
